// DTASNNEmbedding_57638461112386
// MI455X (gfx1250) — matrix operand order repaired and checked by reading, not compiled, not run on hardware
//
#include <hip/hip_runtime.h>
#include <stdint.h>

// ---------------------------------------------------------------------------
// DTA-SNN gated-LIF recurrent conv layer for MI455X (gfx1250, wave32, WMMA).
//
// Both 3x3 convs per step fuse into one implicit GEMM per time step:
//   M = B*H*W = 32768 pixels, N = 128 channels, K = 2*64*9 = 1152
// using V_WMMA_F32_16X16X32_F16. Packed-weight (B-matrix) chunks are staged
// into LDS with GLOBAL_LOAD_ASYNC_TO_LDS_B128 (ASYNCcnt, double-buffered,
// overlapped with WMMA). Each wave owns 2 M-tiles so every B fragment feeds
// two WMMAs; B fragments are additionally double-buffered in registers so DS
// latency hides behind WMMA issue. The gated-LIF update runs in-register.
// ---------------------------------------------------------------------------

typedef __attribute__((ext_vector_type(16))) _Float16 v16h;
typedef __attribute__((ext_vector_type(8)))  _Float16 v8h;
typedef __attribute__((ext_vector_type(8)))  float    v8f;

namespace {
constexpr int BB   = 8;    // batch
constexpr int TT   = 5;    // time steps
constexpr int CIN  = 64;
constexpr int COUT = 64;
constexpr int HH   = 64;
constexpr int WW   = 64;
constexpr int KTOT   = 2 * CIN * 9;    // 1152
constexpr int CHUNKS = KTOT / 32;      // 36 K-chunks of 32
constexpr int CSTR   = 72;             // padded LDS ci-stride (bank spread)

// dynamic LDS partition (bytes)
constexpr unsigned HALO_BYTES = 4 * 66 * CSTR * 2;            // 38016 per path
constexpr unsigned OFF_HALO_S = HALO_BYTES;                   // 38016
constexpr unsigned OFF_BBUF0  = 2 * HALO_BYTES;               // 76032
constexpr unsigned OFF_BBUF1  = OFF_BBUF0 + 8192;             // 84224
constexpr unsigned SMEM_BYTES = OFF_BBUF1 + 8192;             // 92416

constexpr size_t WPK_BYTES  = (size_t)CHUNKS * 8 * 32 * 16 * 2; // 294912 B
constexpr size_t STATE_ELEM = (size_t)BB * COUT * HH * WW;      // 2097152
}

// ---------------------------------------------------------------------------
// Pack [w_in ; w_gate] (f32, OIHW) into B-fragment-native f16 layout:
//   wpk[((chunk*8 + ntile)*32 + lane)*16 + e]
// chunk c -> p = c/18 (0: w_in, 1: w_gate), dydx = (c%18)>>1, ci0 = (c&1)*32
// lane < 16: col n = ntile*16+lane,    elements e -> K = ci0 + e
// lane >=16: col n = ntile*16+lane-16, elements e -> K = ci0 + 16 + e
// ---------------------------------------------------------------------------
__global__ __launch_bounds__(256) void pack_weights_kernel(
    const float* __restrict__ w_in, const float* __restrict__ w_gate,
    _Float16* __restrict__ wpk)
{
    int t = blockIdx.x * 256 + threadIdx.x;     // CHUNKS*8*32*16 = 147456
    if (t >= CHUNKS * 8 * 32 * 16) return;
    int e = t & 15;
    int l = (t >> 4) & 31;
    int j = (t >> 9) & 7;
    int c = t >> 12;
    int p    = c / 18;
    int rem  = c % 18;
    int dydx = rem >> 1;
    int ci0  = (rem & 1) << 5;
    int dy   = dydx / 3;
    int dx   = dydx - dy * 3;
    int n    = j * 16 + (l & 15);
    int kl   = (l < 16) ? e : (e + 16);
    int ci   = ci0 + kl;
    const float* wsrc = (p == 0) ? w_in : w_gate;
    float v = wsrc[(((size_t)n * CIN + ci) * 3 + dy) * 3 + dx];
    wpk[t] = (_Float16)v;
}

// ---------------------------------------------------------------------------
// Zero vmem (f32) and the first spike ping-pong buffer (f16).
// ---------------------------------------------------------------------------
__global__ __launch_bounds__(256) void init_state_kernel(
    float* __restrict__ vmem, _Float16* __restrict__ spike0, size_t n)
{
    size_t i = (size_t)blockIdx.x * 256 + threadIdx.x;
    if (i < n) {
        vmem[i]   = 0.0f;
        spike0[i] = (_Float16)0.0f;
    }
}

// ---------------------------------------------------------------------------
// One recurrent time step. Block = 128 threads (4 waves) -> two output rows
// (128 px); wave w owns pixels [w*32, w*32+32) (2 M-tiles) x all 128 ch.
// ---------------------------------------------------------------------------
__global__ __launch_bounds__(128) void snn_step_kernel(
    const float*    __restrict__ events,    // [B,T,CIN,H,W] f32
    const float*    __restrict__ b_in,      // [128]
    const float*    __restrict__ b_gate,    // [128]
    const float*    __restrict__ tdecay,    // [64]
    const _Float16* __restrict__ wpk,       // packed B fragments
    float*          __restrict__ vmem,      // [B,64,H,W] f32 state
    const _Float16* __restrict__ spike_in,  // [B,64,H,W] f16 (t-1)
    _Float16*       __restrict__ spike_out, // [B,64,H,W] f16 (t)
    float*          __restrict__ out,       // [B,64,H,W] f32 running mean
    int s, int tev, int first)
{
    extern __shared__ _Float16 smem[];
    _Float16* haloE = smem;                               // [4][66][CSTR]
    _Float16* haloS = smem + OFF_HALO_S / 2;

    const int tid = threadIdx.x;
    const int bm  = blockIdx.x;       // 0..255
    const int b   = bm >> 5;          // batch
    const int y0  = (bm & 31) << 1;   // first of two output rows

    // ---- Stage 4-row halo (y0-1..y0+2) x 66 cols x 64 ch, both paths -------
    for (int idx = tid; idx < 4 * 66 * 64; idx += 128) {
        int ci  = idx & 63;
        int cr  = idx >> 6;           // row*66 + col
        int col = cr % 66;
        int row = cr / 66;
        int y   = y0 - 1 + row;
        int x   = col - 1;
        float ev = 0.0f, sp = 0.0f;
        if ((unsigned)y < 64u && (unsigned)x < 64u) {
            ev = events[((((size_t)b * TT + tev) * CIN + ci) * HH + y) * WW + x];
            sp = (float)spike_in[(((size_t)b * COUT + ci) * HH + y) * WW + x];
        }
        haloE[cr * CSTR + ci] = (_Float16)ev;
        haloS[cr * CSTR + ci] = (_Float16)sp;
    }

    // ---- Async-stage B chunk 0 into LDS buffer 0 (ASYNCcnt) ----------------
    const unsigned ldsbase = __builtin_amdgcn_groupstaticsize();
    {
        unsigned lds0 = ldsbase + OFF_BBUF0 + (unsigned)tid * 16u;
        uint64_t g    = (uint64_t)(uintptr_t)wpk + (uint64_t)tid * 16u;
#pragma unroll
        for (int k = 0; k < 4; ++k) {
            asm volatile("global_load_async_to_lds_b128 %0, %1, off"
                         :: "v"(lds0 + (unsigned)k * 2048u),
                            "v"(g + (uint64_t)k * 2048u)
                         : "memory");
        }
    }

    const int wv    = tid >> 5;
    const int lane  = tid & 31;
    const int mrow  = lane & 15;               // A row within 16x16 tile
    const int khalf = (lane < 16) ? 0 : 16;    // A-fragment K half per layout

    v8f zf = {};
    v8f acc[2][8];
#pragma unroll
    for (int mt = 0; mt < 2; ++mt)
#pragma unroll
        for (int j = 0; j < 8; ++j) acc[mt][j] = zf;

    // ---- Implicit GEMM: 36 K-chunks, async double-buffered B ---------------
    int cc = 0;
    for (int p = 0; p < 2; ++p) {
        const _Float16* hsrc = (p == 0) ? haloE : haloS;
        for (int dy = 0; dy < 3; ++dy) {
            for (int dx = 0; dx < 3; ++dx) {
#pragma unroll
                for (int h = 0; h < 2; ++h) {
                    // current buffer's async writes have landed in this wave;
                    // barrier makes them (and prior-chunk reads) block-wide
                    asm volatile("s_wait_asynccnt 0" ::: "memory");
                    __syncthreads();
                    if (cc + 1 < CHUNKS) {
                        unsigned nboff = ((cc + 1) & 1) ? OFF_BBUF1 : OFF_BBUF0;
                        unsigned lds0 = ldsbase + nboff + (unsigned)tid * 16u;
                        uint64_t g = (uint64_t)(uintptr_t)wpk
                                     + (uint64_t)(cc + 1) * 8192u
                                     + (uint64_t)tid * 16u;
#pragma unroll
                        for (int k = 0; k < 4; ++k) {
                            asm volatile(
                                "global_load_async_to_lds_b128 %0, %1, off"
                                :: "v"(lds0 + (unsigned)k * 2048u),
                                   "v"(g + (uint64_t)k * 2048u)
                                : "memory");
                        }
                    }
                    // active B buffer: offset arithmetic (no LDS pointer array)
                    const _Float16* bb =
                        smem + OFF_BBUF0 / 2 + (unsigned)(cc & 1) * 4096u;
                    const int ci0 = h << 5;

                    v16h aF[2];
#pragma unroll
                    for (int mt = 0; mt < 2; ++mt) {
                        int base = wv * 32 + mt * 16;       // 16-aligned
                        int x    = (base & 63) + mrow;
                        int aoff = (((base >> 6) + dy) * 66 + x + dx) * CSTR
                                   + ci0 + khalf;
                        v8h a0 = *(const v8h*)(hsrc + aoff);
                        v8h a1 = *(const v8h*)(hsrc + aoff + 8);
#pragma unroll
                        for (int q = 0; q < 8; ++q) {
                            aF[mt][q] = a0[q]; aF[mt][q + 8] = a1[q];
                        }
                    }

                    // j-loop with B fragment double-buffered in registers so
                    // the in-order DS queue keeps one load pair in flight
                    // behind each WMMA pair (partial s_wait_dscnt).
                    v8h b0 = *(const v8h*)(bb + lane * 16);
                    v8h b1 = *(const v8h*)(bb + lane * 16 + 8);
#pragma unroll
                    for (int j = 0; j < 8; ++j) {
                        v8h n0, n1;
                        if (j + 1 < 8) {
                            n0 = *(const v8h*)(bb + (j + 1) * 512 + lane * 16);
                            n1 = *(const v8h*)(bb + (j + 1) * 512 + lane * 16 + 8);
                        }
                        v16h bF;
#pragma unroll
                        for (int q = 0; q < 8; ++q) {
                            bF[q] = b0[q]; bF[q + 8] = b1[q];
                        }
                        acc[0][j] = __builtin_amdgcn_wmma_f32_16x16x32_f16(
                            false, aF[0], false, bF, (short)0, acc[0][j],
                            false, false);
                        acc[1][j] = __builtin_amdgcn_wmma_f32_16x16x32_f16(
                            false, aF[1], false, bF, (short)0, acc[1][j],
                            false, false);
                        b0 = n0; b1 = n1;
                    }
                    ++cc;
                }
            }
        }
    }

    // ---- Gated-LIF epilogue: tile j = current ch, tile j+4 = gate ch -------
    const float invT = 1.0f / (float)TT;
#pragma unroll
    for (int mt = 0; mt < 2; ++mt) {
        int base = wv * 32 + mt * 16;
        int yy   = y0 + (base >> 6);
        int xb   = base & 63;
#pragma unroll
        for (int j = 0; j < 4; ++j) {
#pragma unroll
            for (int i = 0; i < 8; ++i) {
                int m  = i + ((lane >> 4) << 3);
                int n  = lane & 15;
                int x  = xb + m;
                int ch = j * 16 + n;
                float cur = acc[mt][j][i]     + b_in[ch]      + b_gate[ch];
                float gp  = acc[mt][j + 4][i] + b_in[64 + ch] + b_gate[64 + ch];
                float gate = 1.0f / (1.0f + __expf(-gp));
                float td = tdecay[ch];
                float th = 1.0f;                    // THRESH * td^s
                for (int q = 0; q < s; ++q) th *= td;
                size_t gidx = (((size_t)b * COUT + ch) * HH + yy) * WW + x;
                float v   = gate * vmem[gidx] + cur;
                float spk = (v - th >= 0.0f) ? 1.0f : 0.0f;
                vmem[gidx]      = v * (1.0f - spk);  // hard reset, VRESET = 0
                spike_out[gidx] = (_Float16)spk;
                float prev = first ? 0.0f : out[gidx];
                out[gidx]  = prev + spk * invT;      // running temporal mean
            }
        }
    }
}

// ---------------------------------------------------------------------------
// Workspace layout (requires ~17.2 MB):
//   [0, 294912)   packed weights (f16)
//   +8MB          vmem state (f32)
//   +4MB          spike ping buffer (f16)
//   +4MB          spike pong buffer (f16)
// ---------------------------------------------------------------------------
extern "C" void kernel_launch(void* const* d_in, const int* in_sizes, int n_in,
                              void* d_out, int out_size, void* d_ws, size_t ws_size,
                              hipStream_t stream) {
    (void)in_sizes; (void)n_in; (void)out_size; (void)ws_size;
    const float* events = (const float*)d_in[0];
    const float* w_in   = (const float*)d_in[1];
    const float* b_in   = (const float*)d_in[2];
    const float* w_gate = (const float*)d_in[3];
    const float* b_gate = (const float*)d_in[4];
    const float* tdecay = (const float*)d_in[5];
    float* out = (float*)d_out;

    char* ws = (char*)d_ws;
    _Float16* wpk  = (_Float16*)ws;
    float*    vmem = (float*)(ws + WPK_BYTES);
    _Float16* spk0 = (_Float16*)(ws + WPK_BYTES + STATE_ELEM * sizeof(float));
    _Float16* spk1 = spk0 + STATE_ELEM;

    pack_weights_kernel<<<(CHUNKS * 8 * 32 * 16 + 255) / 256, 256, 0, stream>>>(
        w_in, w_gate, wpk);
    init_state_kernel<<<(int)((STATE_ELEM + 255) / 256), 256, 0, stream>>>(
        vmem, spk0, STATE_ELEM);

    for (int st = 0; st < TT; ++st) {
        const _Float16* sin  = (st & 1) ? spk1 : spk0;
        _Float16*       sout = (st & 1) ? spk0 : spk1;
        snn_step_kernel<<<BB * (HH / 2), 128, SMEM_BYTES, stream>>>(
            events, b_in, b_gate, tdecay, wpk, vmem, sin, sout, out,
            st, TT - 1 - st, (st == 0) ? 1 : 0);
    }
}
